// LayoutRobertaSelfAttention_12378095747382
// MI455X (gfx1250) — compile-verified, matrix operand order checked by reading, not run on hardware
//
#include <hip/hip_runtime.h>
#include <hip/hip_bf16.h>
#include <cstdint>

#define DEVINL __device__ __forceinline__

static constexpr int B_ = 2, S_ = 1024, HID_ = 1024, NH_ = 16, DH_ = 64;

typedef __bf16 bf16_t;
typedef bf16_t bf16x16 __attribute__((ext_vector_type(16)));
typedef float  f32x8  __attribute__((ext_vector_type(8)));
typedef unsigned int u32x4 __attribute__((ext_vector_type(4)));

struct U32x8 { u32x4 lo, hi; };

DEVINL bf16x16 as_frag(u32x4 lo, u32x4 hi) {
  U32x8 t{lo, hi};
  return __builtin_bit_cast(bf16x16, t);
}

DEVINL unsigned short f2bf(float x) {
  unsigned int u = __builtin_bit_cast(unsigned int, x);
  u += 0x7FFFu + ((u >> 16) & 1u);     // round-to-nearest-even
  return (unsigned short)(u >> 16);
}
DEVINL float bf2f(unsigned short b) {
  return __builtin_bit_cast(float, ((unsigned int)b) << 16);
}

// Load a v16bf WMMA fragment (A-style layout, also used for B as [N,K] rows):
// lanes 0-15: row = lane, K = k0+0..7 and k0+16..23
// lanes 16-31: row = lane-16, K = k0+8..15 and k0+24..31
// Caller folds the row offset into rowbase.
DEVINL bf16x16 load_frag(const unsigned short* rowbase, int k0, int hf) {
  const unsigned short* p = rowbase + k0 + hf * 8;
  u32x4 lo = *(const u32x4*)(p);
  u32x4 hi = *(const u32x4*)(p + 16);
  return as_frag(lo, hi);
}

DEVINL f32x8 wmma_bf16(bf16x16 a, bf16x16 b, f32x8 c) {
  // (neg_a, A, neg_b, B, c_mod, C, reuse_a, reuse_b)
  return __builtin_amdgcn_wmma_f32_16x16x32_bf16(false, a, false, b, (short)0, c,
                                                 false, false);
}

DEVINL f32x8 zero8() {
  f32x8 z;
#pragma unroll
  for (int i = 0; i < 8; ++i) z[i] = 0.f;
  return z;
}

DEVINL float rmax16(float x) {
  x = fmaxf(x, __shfl_xor(x, 1));
  x = fmaxf(x, __shfl_xor(x, 2));
  x = fmaxf(x, __shfl_xor(x, 4));
  x = fmaxf(x, __shfl_xor(x, 8));
  return x;
}
DEVINL float rsum16(float x) {
  x += __shfl_xor(x, 1);
  x += __shfl_xor(x, 2);
  x += __shfl_xor(x, 4);
  x += __shfl_xor(x, 8);
  return x;
}

// ---------------- staging kernels ----------------

__global__ __launch_bounds__(256) void cvt_f32_bf16(const float* __restrict__ src,
                                                    unsigned short* __restrict__ dst,
                                                    int srcN, int dstN) {
  int i = blockIdx.x * 256 + threadIdx.x;
  if (i < dstN) dst[i] = (i < srcN) ? f2bf(src[i]) : (unsigned short)0;
}

// bbox_pos_emb [S,B,HID] -> bf16 [B,S,HID]
__global__ __launch_bounds__(256) void permute_pos(const float* __restrict__ src,
                                                   unsigned short* __restrict__ dst) {
  int i = blockIdx.x * 256 + threadIdx.x;
  int tot = B_ * S_ * HID_;
  if (i >= tot) return;
  int c = i & (HID_ - 1);
  int t = i >> 10;            // b*S + s
  int b = t >> 10;
  int s = t & (S_ - 1);
  dst[i] = f2bf(src[((size_t)(s * B_ + b) << 10) + c]);
}

// Abb = (Q + 1) * QP elementwise, bf16
__global__ __launch_bounds__(256) void abb_kernel(const unsigned short* __restrict__ Q,
                                                  const unsigned short* __restrict__ QP,
                                                  unsigned short* __restrict__ A, int n) {
  int i = blockIdx.x * 256 + threadIdx.x;
  if (i < n) A[i] = f2bf((bf2f(Q[i]) + 1.0f) * bf2f(QP[i]));
}

// ---------------- projection GEMM ----------------
// C[t,o] = sum_i X[t,i]*W[o,i] + bias[o];  X:[2048,1024] bf16, W:[1024,1024] bf16
// Per wave: 32(M) x 64(N); WG: 8 waves = 2x4 -> 64x256; grid (4, 32).
// storeVT=0: out[t*1024 + o] bf16;  storeVT=1: out[((b*16+h)*64+d)*1024 + s] bf16
__global__ __launch_bounds__(256) void gemm_proj(const unsigned short* __restrict__ X,
                                                 const unsigned short* __restrict__ Wt,
                                                 const float* __restrict__ bias,
                                                 unsigned short* __restrict__ out,
                                                 int storeVT) {
  const int K = 1024, N = 1024;
  int wave = threadIdx.x >> 5, lane = threadIdx.x & 31;
  int hf = lane >> 4, j = lane & 15;
  int wm = wave >> 2, wn = wave & 3;
  int m0 = blockIdx.y * 64 + wm * 32;
  int n0 = blockIdx.x * 256 + wn * 64;

  f32x8 acc[2][4];
#pragma unroll
  for (int a = 0; a < 2; ++a)
#pragma unroll
    for (int t = 0; t < 4; ++t) acc[a][t] = zero8();

  const unsigned short* xrow0 = X + (size_t)(m0 + j) * K;
  const unsigned short* xrow1 = X + (size_t)(m0 + 16 + j) * K;
  const unsigned short* wrow[4];
#pragma unroll
  for (int t = 0; t < 4; ++t) wrow[t] = Wt + (size_t)(n0 + t * 16 + j) * K;

#pragma unroll 4
  for (int k = 0; k < K; k += 32) {
    bf16x16 a0 = load_frag(xrow0, k, hf);
    bf16x16 a1 = load_frag(xrow1, k, hf);
#pragma unroll
    for (int t = 0; t < 4; ++t) {
      bf16x16 bfr = load_frag(wrow[t], k, hf);
      acc[0][t] = wmma_bf16(a0, bfr, acc[0][t]);
      acc[1][t] = wmma_bf16(a1, bfr, acc[1][t]);
    }
  }

#pragma unroll
  for (int t = 0; t < 4; ++t) {
    float bv = bias[n0 + t * 16 + j];
#pragma unroll
    for (int mt = 0; mt < 2; ++mt) {
#pragma unroll
      for (int v = 0; v < 8; ++v) {
        int M = v + 8 * hf;
        int row = m0 + mt * 16 + M;          // t index (b*S+s)
        int o = n0 + t * 16 + j;             // output feature
        float val = acc[mt][t][v] + bv;
        if (!storeVT) {
          out[(size_t)row * N + o] = f2bf(val);
        } else {
          int h = o >> 6, d = o & 63;
          int bb = row >> 10, s = row & 1023;
          out[((size_t)((bb * NH_ + h) * DH_ + d) << 10) + s] = f2bf(val);
        }
      }
    }
  }
}

// ---------------- fused attention ----------------
// grid (S/128, NH, B), 256 threads. Wave w owns 16 query rows, full 1024 keys,
// flash-style online softmax (base-2).
__global__ __launch_bounds__(256) void attn_kernel(
    const unsigned short* __restrict__ Qb,   // [B,S,HID] bf16
    const unsigned short* __restrict__ Kb,   // [B,S,HID] bf16
    const unsigned short* __restrict__ Ab,   // (q+1)*qp, [B,S,HID] bf16
    const unsigned short* __restrict__ KPb,  // [B,S,HID] bf16
    const unsigned short* __restrict__ VT,   // [B*NH, DH, S] bf16
    const unsigned short* __restrict__ Db,   // dist_emb bf16 [2048,64] (row 2047 pad)
    const int* __restrict__ maskp,           // bbox_pos_mask [B,S] int32
    const float* __restrict__ amask,         // attention_mask [B,1,1,S]
    float* __restrict__ outp)                // [B,S,HID] f32
{
  __shared__ float Tbuf[8 * 16 * 32];          // per-wave relbias tile (fp32)
  __shared__ unsigned short Pbuf[8 * 16 * 32]; // per-wave P tile (bf16)

  int b = blockIdx.z, h = blockIdx.y;
  int wave = threadIdx.x >> 5, lane = threadIdx.x & 31;
  int hf = lane >> 4, j = lane & 15;
  int l0 = blockIdx.x * 128 + wave * 16;
  size_t bh = (size_t)(b * NH_ + h);

  const float kScale = 0.18033688011112042f;  // log2(e)/sqrt(64)
  const float kLog2e = 1.4426950408889634f;

  const unsigned short* qrow = Qb + ((size_t)(b * S_ + l0 + j) << 10) + h * DH_;
  const unsigned short* arow = Ab + ((size_t)(b * S_ + l0 + j) << 10) + h * DH_;
  bf16x16 qa0 = load_frag(qrow, 0, hf), qa1 = load_frag(qrow, 32, hf);
  bf16x16 aa0 = load_frag(arow, 0, hf), aa1 = load_frag(arow, 32, hf);

  float rm[8], rs[8], mr[8];
#pragma unroll
  for (int v = 0; v < 8; ++v) {
    rm[v] = -1e30f;
    rs[v] = 0.f;
    mr[v] = 1.0f - (float)maskp[b * S_ + l0 + v + 8 * hf];
  }
  f32x8 ctx[4];
#pragma unroll
  for (int n = 0; n < 4; ++n) ctx[n] = zero8();

  float* Tw = Tbuf + wave * 512;
  unsigned short* Pw = Pbuf + wave * 512;

  for (int r0 = 0; r0 < S_; r0 += 32) {
    float sc[2][8];
#pragma unroll
    for (int t = 0; t < 2; ++t) {
      int rt = r0 + t * 16;
      const unsigned short* krow = Kb + ((size_t)(b * S_ + rt + j) << 10) + h * DH_;
      const unsigned short* kprow = KPb + ((size_t)(b * S_ + rt + j) << 10) + h * DH_;

      f32x8 acc = zero8();
      acc = wmma_bf16(qa0, load_frag(krow, 0, hf), acc);
      acc = wmma_bf16(qa1, load_frag(krow, 32, hf), acc);

      f32x8 acc2 = zero8();
      acc2 = wmma_bf16(aa0, load_frag(kprow, 0, hf), acc2);
      acc2 = wmma_bf16(aa1, load_frag(kprow, 32, hf), acc2);

      // relative-position bias: T[i,p] = q[l0+i] . dist_emb[base+p], p = i-j+15
      int base = l0 - rt + 1008;
      const unsigned short* drow0 = Db + (size_t)(base + j) * DH_;
      const unsigned short* drow1 = Db + (size_t)(base + 16 + j) * DH_;
      f32x8 t0 = zero8(), t1 = zero8();
      t0 = wmma_bf16(qa0, load_frag(drow0, 0, hf), t0);
      t0 = wmma_bf16(qa1, load_frag(drow0, 32, hf), t0);
      t1 = wmma_bf16(qa0, load_frag(drow1, 0, hf), t1);
      t1 = wmma_bf16(qa1, load_frag(drow1, 32, hf), t1);
#pragma unroll
      for (int v = 0; v < 8; ++v) {
        int M = v + 8 * hf;
        Tw[M * 32 + j] = t0[v];
        Tw[M * 32 + 16 + j] = t1[v];
      }
      asm volatile("s_wait_dscnt 0" ::: "memory");

      float cm = 1.0f - (float)maskp[b * S_ + rt + j];
      float am = amask[b * S_ + rt + j] * kLog2e;
#pragma unroll
      for (int v = 0; v < 8; ++v) {
        int M = v + 8 * hf;
        float rb = Tw[M * 32 + (M - j) + 15];  // diagonal gather, p in [0,30]
        sc[t][v] = (acc[v] + rb + acc2[v] * (mr[v] * cm)) * kScale + am;
      }
    }

    // ---- online softmax update over this 32-key chunk ----
    float nm[8], f[8], psum[8];
#pragma unroll
    for (int v = 0; v < 8; ++v) {
      float tm = rmax16(fmaxf(sc[0][v], sc[1][v]));
      nm[v] = fmaxf(rm[v], tm);
      f[v] = exp2f(rm[v] - nm[v]);
      rm[v] = nm[v];
    }
#pragma unroll
    for (int v = 0; v < 8; ++v) {
      float p0 = exp2f(sc[0][v] - nm[v]);
      float p1 = exp2f(sc[1][v] - nm[v]);
      int M = v + 8 * hf;
      Pw[M * 32 + j] = f2bf(p0);
      Pw[M * 32 + 16 + j] = f2bf(p1);
      psum[v] = p0 + p1;
    }
#pragma unroll
    for (int v = 0; v < 8; ++v) rs[v] = rs[v] * f[v] + rsum16(psum[v]);
#pragma unroll
    for (int n = 0; n < 4; ++n)
#pragma unroll
      for (int v = 0; v < 8; ++v) ctx[n][v] *= f[v];

    asm volatile("s_wait_dscnt 0" ::: "memory");
    bf16x16 pf = load_frag(Pw + j * 32, 0, hf);  // P in A-layout from LDS
#pragma unroll
    for (int n = 0; n < 4; ++n) {
      const unsigned short* vrow = VT + ((bh * DH_ + n * 16 + j) << 10) + r0;
      ctx[n] = wmma_bf16(pf, load_frag(vrow, 0, hf), ctx[n]);
    }
  }

  // ---- finalize: ctx / rowsum, store fp32 [B,S,HID] ----
#pragma unroll
  for (int n = 0; n < 4; ++n) {
#pragma unroll
    for (int v = 0; v < 8; ++v) {
      int M = v + 8 * hf;
      float val = ctx[n][v] / rs[v];
      outp[((size_t)(b * S_ + l0 + M) << 10) + h * DH_ + n * 16 + j] = val;
    }
  }
}

// ---------------- host launcher ----------------

extern "C" void kernel_launch(void* const* d_in, const int* in_sizes, int n_in,
                              void* d_out, int out_size, void* d_ws, size_t ws_size,
                              hipStream_t stream) {
  const float* hs    = (const float*)d_in[0];
  const float* amask = (const float*)d_in[1];
  const float* pos   = (const float*)d_in[2];
  const int*   bmask = (const int*)d_in[3];
  const float* Wq    = (const float*)d_in[4];
  const float* bq    = (const float*)d_in[5];
  const float* Wk    = (const float*)d_in[6];
  const float* bk    = (const float*)d_in[7];
  const float* Wv    = (const float*)d_in[8];
  const float* bv    = (const float*)d_in[9];
  const float* dist  = (const float*)d_in[10];
  float* outp = (float*)d_out;

  char* ws = (char*)d_ws;
  size_t off = 0;
  auto take = [&](size_t elems) {
    unsigned short* p = (unsigned short*)(ws + off);
    off = (off + elems * 2 + 255) & ~(size_t)255;
    return p;
  };
  const size_t nAct = (size_t)B_ * S_ * HID_;  // 2M
  const size_t nW = (size_t)HID_ * HID_;       // 1M
  unsigned short* hsb  = take(nAct);
  unsigned short* posb = take(nAct);
  unsigned short* wqb  = take(nW);
  unsigned short* wkb  = take(nW);
  unsigned short* wvb  = take(nW);
  unsigned short* distb = take((size_t)2048 * 64);
  unsigned short* Qb   = take(nAct);
  unsigned short* Kb   = take(nAct);
  unsigned short* QPb  = take(nAct);
  unsigned short* KPb  = take(nAct);
  unsigned short* Abb  = take(nAct);
  unsigned short* VTb  = take(nAct);
  (void)ws_size; (void)in_sizes; (void)n_in; (void)out_size;

  auto blocks = [](size_t n) { return (unsigned)((n + 255) / 256); };

  // stage everything to bf16
  cvt_f32_bf16<<<blocks(nAct), 256, 0, stream>>>(hs, hsb, (int)nAct, (int)nAct);
  permute_pos<<<blocks(nAct), 256, 0, stream>>>(pos, posb);
  cvt_f32_bf16<<<blocks(nW), 256, 0, stream>>>(Wq, wqb, (int)nW, (int)nW);
  cvt_f32_bf16<<<blocks(nW), 256, 0, stream>>>(Wk, wkb, (int)nW, (int)nW);
  cvt_f32_bf16<<<blocks(nW), 256, 0, stream>>>(Wv, wvb, (int)nW, (int)nW);
  cvt_f32_bf16<<<blocks(2048 * 64), 256, 0, stream>>>(dist, distb, 2047 * 64, 2048 * 64);

  // projections (WMMA GEMMs)
  dim3 gg(4, 32);
  gemm_proj<<<gg, 256, 0, stream>>>(hsb, wqb, bq, Qb, 0);
  gemm_proj<<<gg, 256, 0, stream>>>(hsb, wkb, bk, Kb, 0);
  gemm_proj<<<gg, 256, 0, stream>>>(hsb, wvb, bv, VTb, 1);   // V stored transposed
  gemm_proj<<<gg, 256, 0, stream>>>(posb, wqb, bq, QPb, 0);
  gemm_proj<<<gg, 256, 0, stream>>>(posb, wkb, bk, KPb, 0);

  abb_kernel<<<blocks(nAct), 256, 0, stream>>>(Qb, QPb, Abb, (int)nAct);

  // fused attention
  attn_kernel<<<dim3(S_ / 128, NH_, B_), 256, 0, stream>>>(
      Qb, Kb, Abb, KPb, VTb, distb, bmask, amask, outp);
}
